// BinaryConv2d_9835475107916
// MI455X (gfx1250) — 8-bit matrix operand order repaired and checked by reading, compiled, not run on hardware
//
#include <hip/hip_runtime.h>
#include <hip/hip_bf16.h>
#include <stddef.h>


typedef __attribute__((ext_vector_type(8))) int v8i;

#define BB 16
#define CC 64
#define HH 256
#define WW 256
#define HW (HH * WW)

__global__ void __launch_bounds__(256) binarize_act_kernel(
    const float* __restrict__ x, const float* __restrict__ rd_k,
    const float* __restrict__ rd_b, signed char* __restrict__ act)
{
    __shared__ signed char tile[64 * 68];
    const int bi = blockIdx.x;
    const int wt = bi & 3;
    const int h  = (bi >> 2) & 255;
    const int b  = bi >> 10;
    const int w0 = wt * 64;
    const int tid = threadIdx.x;

#pragma unroll
    for (int i = 0; i < 16; ++i) {
        const int idx = i * 256 + tid;
        const int c = idx >> 6;
        const int w = idx & 63;
        float v = x[(((size_t)b * CC + c) * HH + h) * WW + w0 + w];
        v = v * rd_k[c] + rd_b[c];
        tile[w * 68 + c] = (v > 0.0f) ? (signed char)1
                         : ((v < 0.0f) ? (signed char)-1 : (signed char)0);
    }
    __syncthreads();

    const int w  = tid >> 2;
    const int c0 = (tid & 3) << 4;
    const int* tp = (const int*)(tile + w * 68 + c0);
    int4 q;
    q.x = tp[0]; q.y = tp[1]; q.z = tp[2]; q.w = tp[3];
    *(int4*)(act + ((((size_t)b * HH + h) * WW + w0 + w) << 6) + c0) = q;
}

__global__ void weight_scale_kernel(const float* __restrict__ cw,
                                    float* __restrict__ scaling)
{
    const int c = threadIdx.x;
    float s = 0.0f;
    for (int i = 0; i < 576; ++i) s += fabsf(cw[c * 576 + i]);
    scaling[c] = s * (1.0f / 576.0f);
}

__global__ void weight_binarize_kernel(const float* __restrict__ cw,
                                       signed char* __restrict__ bfrag)
{
    const int j = blockIdx.x;
    const int kh = j / 3, kw = j % 3;
    const int t = threadIdx.x;
    const int nt = t >> 5, lane = t & 31;
    const int n = nt * 16 + (lane & 15);
    const int hsel = (lane >> 4) * 16;

    union { signed char bch[32]; int4 q[2]; } u;
#pragma unroll
    for (int d = 0; d < 8; ++d) {
#pragma unroll
        for (int bb = 0; bb < 4; ++bb) {
            const int K = (d >> 1) * 16 + (hsel >> 1) + (d & 1) * 4 + bb;
            const float wv = cw[((n * 64 + K) * 3 + kh) * 3 + kw];
            u.bch[d * 4 + bb] = (wv > 0.0f) ? (signed char)1
                              : ((wv < 0.0f) ? (signed char)-1 : (signed char)0);
        }
    }
    int4* dst = (int4*)(bfrag + (((size_t)(j * 4 + nt) * 32 + lane) << 5));
    dst[0] = u.q[0];
    dst[1] = u.q[1];
}

__global__ void __launch_bounds__(256, 2) bconv_wmma_kernel(
    const signed char* __restrict__ act, const signed char* __restrict__ bfrag_g,
    const float* __restrict__ scaling, const float* __restrict__ x,
    const float* __restrict__ pr_bias0, const float* __restrict__ prelu_w,
    const float* __restrict__ pr_bias1, float* __restrict__ out)
{
    __shared__ int4 lds_b[2304];

    const int tid = threadIdx.x;
    const int4* bg = (const int4*)bfrag_g;
#pragma unroll
    for (int i = 0; i < 9; ++i) lds_b[i * 256 + tid] = bg[i * 256 + tid];
    __syncthreads();

    const int b = blockIdx.x >> 8;
    const int h = blockIdx.x & 255;
    const int wave = tid >> 5, lane = tid & 31;
    const int half = lane >> 4, mlo = lane & 15;

    float sc[4], pb0[4], pw[4], pb1[4];
#pragma unroll
    for (int nt = 0; nt < 4; ++nt) {
        const int n = nt * 16 + mlo;
        sc[nt]  = scaling[n];
        pb0[nt] = pr_bias0[n];
        pw[nt]  = prelu_w[n];
        pb1[nt] = pr_bias1[n];
    }

    for (int t = wave * 2; t < wave * 2 + 2; ++t) {
        const int w0 = t * 16;
        const v8i vz = {0, 0, 0, 0, 0, 0, 0, 0};
        v8i acc[4] = {vz, vz, vz, vz};

#pragma unroll
        for (int j = 0; j < 9; ++j) {
            const int kh = j / 3, kw = j % 3;
            const int hh = h + kh - 1;
            const int ww = w0 + mlo + kw - 1;
            const bool valid = (hh >= 0) & (hh < HH) & (ww >= 0) & (ww < WW);
            const int hc = valid ? hh : 0;
            const int wc = valid ? ww : 0;
            const signed char* ap =
                act + ((((size_t)b * HH + hc) * WW + wc) << 6) + (half << 3);
            union { unsigned long long d[4]; v8i v; } A;
#pragma unroll
            for (int d2 = 0; d2 < 4; ++d2) {
                unsigned long long vv = *(const unsigned long long*)(ap + d2 * 16);
                A.d[d2] = valid ? vv : 0ull;
            }
#pragma unroll
            for (int nt = 0; nt < 4; ++nt) {
                union { int4 q[2]; v8i v; } Bf;
                Bf.q[0] = lds_b[(j * 4 + nt) * 64 + lane * 2];
                Bf.q[1] = lds_b[(j * 4 + nt) * 64 + lane * 2 + 1];
                acc[nt] = __builtin_amdgcn_wmma_i32_16x16x64_iu8(
                    true, A.v, true, Bf.v, acc[nt], false, false);
            }
        }

#pragma unroll
        for (int nt = 0; nt < 4; ++nt) {
            const int n = nt * 16 + mlo;
            const size_t base =
                (((size_t)b * CC + n) * HH + h) * WW + w0 + half * 8;
            const float4 xa = *(const float4*)(x + base);
            const float4 xb = *(const float4*)(x + base + 4);
            float o[8];
#pragma unroll
            for (int r = 0; r < 8; ++r) {
                float f = sc[nt] * (float)acc[nt][r] + pb0[nt];
                f = (f >= 0.0f) ? f : f * pw[nt];
                o[r] = f + pb1[nt];
            }
            o[0] += xa.x; o[1] += xa.y; o[2] += xa.z; o[3] += xa.w;
            o[4] += xb.x; o[5] += xb.y; o[6] += xb.z; o[7] += xb.w;
            *(float4*)(out + base)     = make_float4(o[0], o[1], o[2], o[3]);
            *(float4*)(out + base + 4) = make_float4(o[4], o[5], o[6], o[7]);
        }
    }
}

extern "C" void kernel_launch(void* const* d_in, const int* in_sizes, int n_in,
                              void* d_out, int out_size, void* d_ws, size_t ws_size,
                              hipStream_t stream)
{
    const float* x        = (const float*)d_in[0];
    const float* rd_k     = (const float*)d_in[1];
    const float* rd_b     = (const float*)d_in[2];
    const float* conv_w   = (const float*)d_in[4];
    const float* pr_bias0 = (const float*)d_in[5];
    const float* prelu_w  = (const float*)d_in[6];
    const float* pr_bias1 = (const float*)d_in[7];
    float* out = (float*)d_out;

    char* ws = (char*)d_ws;
    signed char* bfrag   = (signed char*)ws;
    float*       scaling = (float*)(ws + 36864);
    signed char* act     = (signed char*)(ws + 65536);

    binarize_act_kernel<<<BB * HH * (WW / 64), 256, 0, stream>>>(x, rd_k, rd_b, act);
    weight_scale_kernel<<<1, 64, 0, stream>>>(conv_w, scaling);
    weight_binarize_kernel<<<9, 128, 0, stream>>>(conv_w, bfrag);
    bconv_wmma_kernel<<<BB * HH, 256, 0, stream>>>(act, bfrag, scaling, x,
                                                   pr_bias0, prelu_w, pr_bias1, out);
}
